// Decoder_4982162063952
// MI455X (gfx1250) — matrix operand order repaired and checked by reading, not compiled, not run on hardware
//
#include <hip/hip_runtime.h>

typedef __attribute__((ext_vector_type(2)))  float    v2f;
typedef __attribute__((ext_vector_type(8)))  float    v8f;
typedef __attribute__((ext_vector_type(8)))  _Float16 v8h;
typedef __attribute__((ext_vector_type(16))) _Float16 v16h;

#define BZ    32      // batch
#define HD    256     // hidden
#define NLAY  2
#define NS    64      // decode steps
#define SENC  64
#define H2    512
#define H3    768
#define VOCAB 32000
#define SOUT  65

__device__ __forceinline__ v8f wmma_f32(v2f a, v2f b, v8f c) {
  return __builtin_amdgcn_wmma_f32_16x16x4_f32(false, a, false, b, (short)0, c, false, false);
}
__device__ __forceinline__ v8f wmma_f16(v16h a, v16h b, v8f c) {
  return __builtin_amdgcn_wmma_f32_16x16x32_f16(false, a, false, b, (short)0, c, false, false);
}
__device__ __forceinline__ float sigmoid_(float v) { return 1.0f / (1.0f + __expf(-v)); }

// ---------------------------------------------------------------------------
// Kernel 1: sequential 2-layer GRU over 64 steps, single workgroup.
// x (32x256) and h (2x32x256) live in LDS; gate GEMMs use f32 WMMA 16x16x4.
// Writes layer-2 hidden per step to Xall (S, B, H) f32.
// ---------------------------------------------------------------------------
__global__ __launch_bounds__(256)
void gru_kernel(const float* __restrict__ enc_hid,
                const int*   __restrict__ words,
                const float* __restrict__ emb,
                const float* __restrict__ W_ih,
                const float* __restrict__ W_hh,
                const float* __restrict__ b_ih,
                const float* __restrict__ b_hh,
                float* __restrict__ Xall) {
  extern __shared__ float smem[];
  float* xs = smem;           // BZ*HD
  float* hs = smem + BZ * HD; // NLAY*BZ*HD
  const int tid   = threadIdx.x;
  const int lane  = tid & 31;
  const int wave  = tid >> 5;
  const int col   = lane & 15;
  const int khalf = lane >> 4;
  const v8f zero = {};

  // h0 = sum of fwd/bwd encoder final hiddens: eh(2,L,B,H)
  for (int i = tid; i < NLAY * BZ * HD; i += 256)
    hs[i] = enc_hid[i] + enc_hid[NLAY * BZ * HD + i];
  __syncthreads();

  for (int t = 0; t < NS; ++t) {
    // x = relu(emb[tok])
    for (int i = tid; i < BZ * HD; i += 256) {
      int b = i >> 8, k = i & 255;
      int tok = words[b * SOUT + t];
      float v = emb[tok * HD + k];
      xs[i] = v > 0.f ? v : 0.f;
    }
    __syncthreads();

    for (int l = 0; l < NLAY; ++l) {
      const float* Wi = W_ih + l * H3 * HD;
      const float* Wh = W_hh + l * H3 * HD;
      float* hl = hs + l * BZ * HD;
      float hnew[2][2][8];  // [ngroup][mtile][row]

      for (int g = 0; g < 2; ++g) {
        const int n0 = (wave * 2 + g) * 16;  // 16 hidden columns owned by this pass
        v8f acc[2][3][2];                    // [mtile][gate][x-path / h-path]
        #pragma unroll
        for (int mt = 0; mt < 2; ++mt)
          #pragma unroll
          for (int gg = 0; gg < 3; ++gg) { acc[mt][gg][0] = zero; acc[mt][gg][1] = zero; }

        for (int k0 = 0; k0 < HD; k0 += 4) {
          v2f ax[2], ah[2];
          #pragma unroll
          for (int mt = 0; mt < 2; ++mt) {
            ax[mt] = *(const v2f*)&xs[(mt * 16 + col) * HD + k0 + 2 * khalf];
            ah[mt] = *(const v2f*)&hl[(mt * 16 + col) * HD + k0 + 2 * khalf];
          }
          #pragma unroll
          for (int gg = 0; gg < 3; ++gg) {
            const int j = gg * HD + n0 + col;  // row of W (3H x H), B-frag = W^T
            v2f bi = *(const v2f*)&Wi[j * HD + k0 + 2 * khalf];
            v2f bh = *(const v2f*)&Wh[j * HD + k0 + 2 * khalf];
            #pragma unroll
            for (int mt = 0; mt < 2; ++mt) {
              acc[mt][gg][0] = wmma_f32(ax[mt], bi, acc[mt][gg][0]);
              acc[mt][gg][1] = wmma_f32(ah[mt], bh, acc[mt][gg][1]);
            }
          }
        }

        const int j0 = n0 + col;
        const float bir = b_ih[l * H3 + 0 * HD + j0], bhr = b_hh[l * H3 + 0 * HD + j0];
        const float biz = b_ih[l * H3 + 1 * HD + j0], bhz = b_hh[l * H3 + 1 * HD + j0];
        const float bin = b_ih[l * H3 + 2 * HD + j0], bhn = b_hh[l * H3 + 2 * HD + j0];
        #pragma unroll
        for (int mt = 0; mt < 2; ++mt) {
          #pragma unroll
          for (int r = 0; r < 8; ++r) {
            const int bb = mt * 16 + 8 * khalf + r;  // batch row per C-layout
            float ir = acc[mt][0][0][r] + bir, hr = acc[mt][0][1][r] + bhr;
            float iz = acc[mt][1][0][r] + biz, hz = acc[mt][1][1][r] + bhz;
            float in_ = acc[mt][2][0][r] + bin, hn = acc[mt][2][1][r] + bhn;
            float rg = sigmoid_(ir + hr);
            float zg = sigmoid_(iz + hz);
            float ng = tanhf(in_ + rg * hn);
            float hp = hl[bb * HD + j0];
            hnew[g][mt][r] = (1.f - zg) * ng + zg * hp;
          }
        }
      }
      __syncthreads();  // everyone done reading old x/h
      #pragma unroll
      for (int g = 0; g < 2; ++g) {
        const int j0 = (wave * 2 + g) * 16 + col;
        #pragma unroll
        for (int mt = 0; mt < 2; ++mt)
          #pragma unroll
          for (int r = 0; r < 8; ++r) {
            const int bb = mt * 16 + 8 * khalf + r;
            float v = hnew[g][mt][r];
            hl[bb * HD + j0] = v;
            xs[bb * HD + j0] = v;  // next layer's input
            if (l == NLAY - 1) Xall[(t * BZ + bb) * HD + j0] = v;
          }
      }
      __syncthreads();
    }
  }
}

// ---------------------------------------------------------------------------
// Kernel 2: per-step attention + pre-logit projection. One block per t.
// q GEMM and O GEMM via f32 WMMA; O stored as f16 for the big vocab GEMM.
// ---------------------------------------------------------------------------
__global__ __launch_bounds__(256)
void attn_kernel(const float* __restrict__ Xall,
                 const float* __restrict__ enc,
                 const float* __restrict__ Wa, const float* __restrict__ ba,
                 const float* __restrict__ Ww, const float* __restrict__ bw,
                 _Float16* __restrict__ Of16) {
  extern __shared__ float smem[];
  float* cat = smem;            // BZ*H3  (cols 0..255 = x, 256..767 = ctx)
  float* q   = smem + BZ * H3;  // BZ*H2
  float* sc  = q + BZ * H2;     // BZ*SENC
  const int t = blockIdx.x;
  const int tid = threadIdx.x;
  const int lane = tid & 31, wave = tid >> 5;
  const int col = lane & 15, khalf = lane >> 4;
  const v8f zero = {};

  for (int i = tid; i < BZ * HD; i += 256) {
    int b = i >> 8, k = i & 255;
    cat[b * H3 + k] = Xall[(t * BZ + b) * HD + k];
  }
  __syncthreads();

  // q = x @ Wa^T + ba : M=32, N=512, K=256
  for (int nt = 0; nt < 4; ++nt) {
    const int n0 = (wave * 4 + nt) * 16;
    v8f acc0 = zero, acc1 = zero;
    for (int k0 = 0; k0 < HD; k0 += 4) {
      v2f a0 = *(const v2f*)&cat[(col) * H3 + k0 + 2 * khalf];
      v2f a1 = *(const v2f*)&cat[(16 + col) * H3 + k0 + 2 * khalf];
      v2f bf = *(const v2f*)&Wa[(n0 + col) * HD + k0 + 2 * khalf];
      acc0 = wmma_f32(a0, bf, acc0);
      acc1 = wmma_f32(a1, bf, acc1);
    }
    const int j0 = n0 + col;
    const float bav = ba[j0];
    #pragma unroll
    for (int r = 0; r < 8; ++r) {
      q[(8 * khalf + r) * H2 + j0]      = acc0[r] + bav;
      q[(16 + 8 * khalf + r) * H2 + j0] = acc1[r] + bav;
    }
  }
  __syncthreads();

  // scores[b][s] = <enc[b][s], q[b]>
  {
    const int b = tid >> 3, s0 = (tid & 7) * 8;
    for (int s = s0; s < s0 + 8; ++s) {
      const float* e = enc + (b * SENC + s) * H2;
      const float* qq = q + b * H2;
      float d = 0.f;
      for (int k = 0; k < H2; ++k) d += e[k] * qq[k];
      sc[b * SENC + s] = d;
    }
  }
  __syncthreads();

  // softmax over s, per batch row
  if (tid < BZ) {
    float m = -1e30f;
    for (int s = 0; s < SENC; ++s) m = fmaxf(m, sc[tid * SENC + s]);
    float sum = 0.f;
    for (int s = 0; s < SENC; ++s) { float e = __expf(sc[tid * SENC + s] - m); sc[tid * SENC + s] = e; sum += e; }
    const float inv = 1.f / sum;
    for (int s = 0; s < SENC; ++s) sc[tid * SENC + s] *= inv;
  }
  __syncthreads();

  // ctx[b][d] = sum_s w[b][s] * enc[b][s][d] -> cat[:,256:768]
  {
    const int b = tid >> 3, d0 = (tid & 7) * 64;
    for (int d = d0; d < d0 + 64; ++d) {
      float a = 0.f;
      for (int s = 0; s < SENC; ++s) a += sc[b * SENC + s] * enc[(b * SENC + s) * H2 + d];
      cat[b * H3 + HD + d] = a;
    }
  }
  __syncthreads();

  // O = relu(cat @ Ww^T + bw): M=32, N=768, K=768; store f16
  for (int nt = 0; nt < 6; ++nt) {
    const int n0 = (wave * 6 + nt) * 16;
    v8f acc0 = zero, acc1 = zero;
    for (int k0 = 0; k0 < H3; k0 += 4) {
      v2f a0 = *(const v2f*)&cat[(col) * H3 + k0 + 2 * khalf];
      v2f a1 = *(const v2f*)&cat[(16 + col) * H3 + k0 + 2 * khalf];
      v2f bf = *(const v2f*)&Ww[(n0 + col) * H3 + k0 + 2 * khalf];
      acc0 = wmma_f32(a0, bf, acc0);
      acc1 = wmma_f32(a1, bf, acc1);
    }
    const int j0 = n0 + col;
    const float bwv = bw[j0];
    #pragma unroll
    for (int r = 0; r < 8; ++r) {
      float o0 = acc0[r] + bwv; o0 = o0 > 0.f ? o0 : 0.f;
      float o1 = acc1[r] + bwv; o1 = o1 > 0.f ? o1 : 0.f;
      Of16[(t * BZ + 8 * khalf + r) * H3 + j0]      = (_Float16)o0;
      Of16[(t * BZ + 16 + 8 * khalf + r) * H3 + j0] = (_Float16)o1;
    }
  }
}

// ---------------------------------------------------------------------------
// Wo f32 -> f16 conversion (one pass; Wo16 stays hot in L2 for kernel 3).
// ---------------------------------------------------------------------------
__global__ __launch_bounds__(256)
void convert_wo_kernel(const float* __restrict__ Wo, _Float16* __restrict__ Wo16) {
  const size_t i0 = ((size_t)blockIdx.x * 256 + threadIdx.x) * 8;
  #pragma unroll
  for (int i = 0; i < 8; ++i) Wo16[i0 + i] = (_Float16)Wo[i0 + i];
}

// ---------------------------------------------------------------------------
// Kernel 3a (preferred): logits = O @ Wo^T + bo. M=2048, N=32000, K=768.
// f16 WMMA 16x16x32, f32 accumulate. 128x128 tile per block, 8 waves (2x4).
// A/B tiles staged with GLOBAL_LOAD_ASYNC_TO_LDS_B128, double buffered,
// pipelined with s_wait_asynccnt (each thread issues exactly 4 per chunk).
// ---------------------------------------------------------------------------
__global__ __launch_bounds__(256)
void logits_async_kernel(const _Float16* __restrict__ Og,
                         const _Float16* __restrict__ Wo16,
                         const float* __restrict__ bo,
                         float* __restrict__ out) {
  __shared__ __align__(16) _Float16 As[2][128 * 32];
  __shared__ __align__(16) _Float16 Bs[2][128 * 32];
  const int n0 = blockIdx.x * 128;
  const int m0 = blockIdx.y * 128;
  const int tid = threadIdx.x;
  const int lane = tid & 31, wave = tid >> 5;
  const int col = lane & 15, khalf = lane >> 4;
  const int wm = wave >> 2, wn = wave & 3;

  const v8f zero = {};
  v8f acc[4][2];
  #pragma unroll
  for (int mt = 0; mt < 4; ++mt) { acc[mt][0] = zero; acc[mt][1] = zero; }

  const int srow = tid >> 1;          // staging row (0..127)
  const int sseg = (tid & 1) * 16;    // staging segment (halves)

  // Issue this thread's 4 async b128 copies (A:2 + B:2) for chunk kc -> buf.
  auto issue_chunk = [&](int kc, int buf) {
    unsigned ldsA = (unsigned)(unsigned long long)&As[buf][srow * 32 + sseg];
    unsigned ldsB = (unsigned)(unsigned long long)&Bs[buf][srow * 32 + sseg];
    unsigned voA = (unsigned)(((m0 + srow) * H3 + kc + sseg) * 2);
    unsigned voB = (unsigned)(((n0 + srow) * H3 + kc + sseg) * 2);
    asm volatile("global_load_async_to_lds_b128 %0, %1, %2"
                 :: "v"(ldsA), "v"(voA), "s"(Og) : "memory");
    asm volatile("global_load_async_to_lds_b128 %0, %1, %2 offset:16"
                 :: "v"(ldsA), "v"(voA), "s"(Og) : "memory");
    asm volatile("global_load_async_to_lds_b128 %0, %1, %2"
                 :: "v"(ldsB), "v"(voB), "s"(Wo16) : "memory");
    asm volatile("global_load_async_to_lds_b128 %0, %1, %2 offset:16"
                 :: "v"(ldsB), "v"(voB), "s"(Wo16) : "memory");
  };

  const int NCH = H3 / 32;  // 24 K-chunks
  issue_chunk(0, 0);
  for (int ci = 0; ci < NCH; ++ci) {
    const int cb = ci & 1;
    if (ci + 1 < NCH) {
      issue_chunk((ci + 1) * 32, cb ^ 1);  // prefetch into other buffer
      asm volatile("s_wait_asynccnt 0x4" ::: "memory");  // chunk ci landed
    } else {
      asm volatile("s_wait_asynccnt 0x0" ::: "memory");
    }
    __syncthreads();  // all waves' chunk-ci data visible in LDS

    v16h afr[4], bfr[2];
    #pragma unroll
    for (int mt = 0; mt < 4; ++mt) {  // A: lane<16 -> K {0-7,16-23}; lane>=16 -> {8-15,24-31}
      const _Float16* p = &As[cb][(wm * 64 + mt * 16 + col) * 32 + khalf * 8];
      v8h lo = *(const v8h*)p;
      v8h hi = *(const v8h*)(p + 16);
      #pragma unroll
      for (int i = 0; i < 8; ++i) { afr[mt][i] = lo[i]; afr[mt][i + 8] = hi[i]; }
    }
    #pragma unroll
    for (int nt = 0; nt < 2; ++nt) {  // B: lane<16 -> K {0-7,16-23}; lane>=16 -> {8-15,24-31}
      const _Float16* p = &Bs[cb][(wn * 32 + nt * 16 + col) * 32 + khalf * 8];
      v8h lo = *(const v8h*)p;
      v8h hi = *(const v8h*)(p + 16);
      #pragma unroll
      for (int i = 0; i < 8; ++i) { bfr[nt][i] = lo[i]; bfr[nt][i + 8] = hi[i]; }
    }
    #pragma unroll
    for (int mt = 0; mt < 4; ++mt)
      #pragma unroll
      for (int nt = 0; nt < 2; ++nt)
        acc[mt][nt] = wmma_f16(afr[mt], bfr[nt], acc[mt][nt]);
    __syncthreads();  // reads of buf cb done before it is refilled (iter ci+1 issues chunk ci+2 into cb)
  }

  // bias + store: row = t*32 + b, out layout (B, S, V)
  #pragma unroll
  for (int nt = 0; nt < 2; ++nt) {
    const int n = n0 + wn * 32 + nt * 16 + col;
    const float bias = bo[n];
    #pragma unroll
    for (int mt = 0; mt < 4; ++mt) {
      #pragma unroll
      for (int r = 0; r < 8; ++r) {
        const int grow = m0 + wm * 64 + mt * 16 + 8 * khalf + r;
        const int b = grow & 31, tt = grow >> 5;
        out[(size_t)(b * NS + tt) * VOCAB + n] = acc[mt][nt][r] + bias;
      }
    }
  }
}

// ---------------------------------------------------------------------------
// Kernel 3b (fallback when workspace can't hold Wo16): same GEMM, synchronous
// staging with on-the-fly f32->f16 conversion of Wo.
// ---------------------------------------------------------------------------
__global__ __launch_bounds__(256)
void logits_kernel(const _Float16* __restrict__ Og,
                   const float* __restrict__ Wo,
                   const float* __restrict__ bo,
                   float* __restrict__ out) {
  __shared__ __align__(16) _Float16 As[128 * 32];
  __shared__ __align__(16) _Float16 Bs[128 * 32];
  const int n0 = blockIdx.x * 128;
  const int m0 = blockIdx.y * 128;
  const int tid = threadIdx.x;
  const int lane = tid & 31, wave = tid >> 5;
  const int col = lane & 15, khalf = lane >> 4;
  const int wm = wave >> 2, wn = wave & 3;

  const v8f zero = {};
  v8f acc[4][2];
  #pragma unroll
  for (int mt = 0; mt < 4; ++mt) { acc[mt][0] = zero; acc[mt][1] = zero; }

  const int srow = tid >> 1;
  const int sseg = (tid & 1) * 16;

  for (int kc = 0; kc < H3; kc += 32) {
    {
      const _Float16* src = Og + (size_t)(m0 + srow) * H3 + kc + sseg;
      *(v8h*)&As[srow * 32 + sseg]     = *(const v8h*)src;
      *(v8h*)&As[srow * 32 + sseg + 8] = *(const v8h*)(src + 8);
    }
    {
      const float* src = Wo + (size_t)(n0 + srow) * H3 + kc + sseg;
      #pragma unroll
      for (int i = 0; i < 16; ++i) Bs[srow * 32 + sseg + i] = (_Float16)src[i];
    }
    __syncthreads();

    v16h afr[4], bfr[2];
    #pragma unroll
    for (int mt = 0; mt < 4; ++mt) {
      const _Float16* p = As + (wm * 64 + mt * 16 + col) * 32 + khalf * 8;
      v8h lo = *(const v8h*)p;
      v8h hi = *(const v8h*)(p + 16);
      #pragma unroll
      for (int i = 0; i < 8; ++i) { afr[mt][i] = lo[i]; afr[mt][i + 8] = hi[i]; }
    }
    #pragma unroll
    for (int nt = 0; nt < 2; ++nt) {
      const _Float16* p = Bs + (wn * 32 + nt * 16 + col) * 32 + khalf * 8;
      v8h lo = *(const v8h*)p;
      v8h hi = *(const v8h*)(p + 16);
      #pragma unroll
      for (int i = 0; i < 8; ++i) { bfr[nt][i] = lo[i]; bfr[nt][i + 8] = hi[i]; }
    }
    #pragma unroll
    for (int mt = 0; mt < 4; ++mt)
      #pragma unroll
      for (int nt = 0; nt < 2; ++nt)
        acc[mt][nt] = wmma_f16(afr[mt], bfr[nt], acc[mt][nt]);
    __syncthreads();
  }

  #pragma unroll
  for (int nt = 0; nt < 2; ++nt) {
    const int n = n0 + wn * 32 + nt * 16 + col;
    const float bias = bo[n];
    #pragma unroll
    for (int mt = 0; mt < 4; ++mt) {
      #pragma unroll
      for (int r = 0; r < 8; ++r) {
        const int grow = m0 + wm * 64 + mt * 16 + 8 * khalf + r;
        const int b = grow & 31, tt = grow >> 5;
        out[(size_t)(b * NS + tt) * VOCAB + n] = acc[mt][nt][r] + bias;
      }
    }
  }
}

extern "C" void kernel_launch(void* const* d_in, const int* in_sizes, int n_in,
                              void* d_out, int out_size, void* d_ws, size_t ws_size,
                              hipStream_t stream) {
  (void)in_sizes; (void)n_in; (void)out_size;
  const float* enc_out = (const float*)d_in[0];
  const float* enc_hid = (const float*)d_in[1];
  const int*   words   = (const int*)  d_in[2];
  const float* emb     = (const float*)d_in[3];
  const float* W_ih    = (const float*)d_in[4];
  const float* W_hh    = (const float*)d_in[5];
  const float* b_ih    = (const float*)d_in[6];
  const float* b_hh    = (const float*)d_in[7];
  const float* Wa      = (const float*)d_in[8];
  const float* ba      = (const float*)d_in[9];
  const float* Ww      = (const float*)d_in[10];
  const float* bw      = (const float*)d_in[11];
  const float* Wo      = (const float*)d_in[12];
  const float* bo      = (const float*)d_in[13];
  float* out = (float*)d_out;

  char* ws = (char*)d_ws;
  const size_t xall_b = (size_t)NS * BZ * HD * sizeof(float);        // 2 MB
  const size_t of16_b = (size_t)NS * BZ * H3 * sizeof(_Float16);     // 3 MB
  const size_t wo16_b = (size_t)VOCAB * H3 * sizeof(_Float16);       // 49.15 MB
  float*    Xall = (float*)ws;
  _Float16* Of16 = (_Float16*)(ws + xall_b);
  _Float16* Wo16 = (_Float16*)(ws + xall_b + of16_b);
  const bool use_async = ws_size >= xall_b + of16_b + wo16_b;        // deterministic

  const size_t lds1 = (size_t)(BZ * HD + NLAY * BZ * HD) * sizeof(float);      // 96 KB
  const size_t lds2 = (size_t)(BZ * H3 + BZ * H2 + BZ * SENC) * sizeof(float); // 168 KB

  if (use_async)
    hipLaunchKernelGGL(convert_wo_kernel, dim3((VOCAB * H3) / (256 * 8)), dim3(256), 0, stream,
                       Wo, Wo16);
  hipLaunchKernelGGL(gru_kernel, dim3(1), dim3(256), lds1, stream,
                     enc_hid, words, emb, W_ih, W_hh, b_ih, b_hh, Xall);
  hipLaunchKernelGGL(attn_kernel, dim3(NS), dim3(256), lds2, stream,
                     Xall, enc_out, Wa, ba, Ww, bw, Of16);
  if (use_async)
    hipLaunchKernelGGL(logits_async_kernel, dim3(VOCAB / 128, (NS * BZ) / 128), dim3(256), 0, stream,
                       Of16, Wo16, bo, out);
  else
    hipLaunchKernelGGL(logits_kernel, dim3(VOCAB / 128, (NS * BZ) / 128), dim3(256), 0, stream,
                       Of16, Wo, bo, out);
}
